// IdentityAwareContrastiveLoss_62886911148102
// MI455X (gfx1250) — matrix operand order repaired and checked by reading, not compiled, not run on hardware
//
#include <hip/hip_runtime.h>
#include <hip/hip_bf16.h>

typedef __bf16 bf16_t;
typedef bf16_t v16bf __attribute__((ext_vector_type(16)));
typedef bf16_t v8bf  __attribute__((ext_vector_type(8)));
typedef float  v8f   __attribute__((ext_vector_type(8)));

#define B_N   4096
#define D_K   2048
#define INV_T (1.0f / 0.07f)
#define MARGIN 0.5f
#define EPSV  1e-6f

// ---------------------------------------------------------------------------
// K1: row norms of face/body, exact f32 diagonal dot, zero accumulators
// ---------------------------------------------------------------------------
__global__ __launch_bounds__(256)
void k_prep(const float* __restrict__ F, const float* __restrict__ Bd,
            float* __restrict__ inv_f, float* __restrict__ inv_b,
            float* __restrict__ diag, float* __restrict__ rowsum,
            float* __restrict__ colsum, float* __restrict__ scal) {
    const int i = blockIdx.x;
    const int t = threadIdx.x;
    const float* fr = F  + (size_t)i * D_K;
    const float* br = Bd + (size_t)i * D_K;
    float sf = 0.f, sb = 0.f, sd = 0.f;
    for (int k = t * 4; k < D_K; k += 1024) {
        float4 f4 = *(const float4*)(fr + k);
        float4 b4 = *(const float4*)(br + k);
        sf += f4.x*f4.x + f4.y*f4.y + f4.z*f4.z + f4.w*f4.w;
        sb += b4.x*b4.x + b4.y*b4.y + b4.z*b4.z + b4.w*b4.w;
        sd += f4.x*b4.x + f4.y*b4.y + f4.z*b4.z + f4.w*b4.w;
    }
    __shared__ float r0[256], r1[256], r2[256];
    r0[t] = sf; r1[t] = sb; r2[t] = sd;
    __syncthreads();
    for (int s = 128; s > 0; s >>= 1) {
        if (t < s) { r0[t] += r0[t+s]; r1[t] += r1[t+s]; r2[t] += r2[t+s]; }
        __syncthreads();
    }
    if (t == 0) {
        float invf = 1.0f / sqrtf(r0[0]);
        float invb = 1.0f / sqrtf(r1[0]);
        inv_f[i] = invf;
        inv_b[i] = invb;
        diag[i]  = r2[0] * invf * invb;   // cosine sim on diagonal (exact f32)
        rowsum[i] = 0.f;
        colsum[i] = 0.f;
        if (i == 0) { scal[0] = 0.f; scal[1] = 0.f; }
    }
}

// ---------------------------------------------------------------------------
// K2: tiled bf16 WMMA GEMM of normalized features + online exp-sum reduction
// ---------------------------------------------------------------------------
struct F16x { float4 a, b, c, d; };

__device__ inline F16x ld16(const float* __restrict__ p) {
    F16x r;
    r.a = *(const float4*)(p);
    r.b = *(const float4*)(p + 4);
    r.c = *(const float4*)(p + 8);
    r.d = *(const float4*)(p + 12);
    return r;
}

__device__ inline void st16(bf16_t* dst, const F16x& x, float s) {
    v8bf lo, hi;
    lo[0]=(bf16_t)(x.a.x*s); lo[1]=(bf16_t)(x.a.y*s); lo[2]=(bf16_t)(x.a.z*s); lo[3]=(bf16_t)(x.a.w*s);
    lo[4]=(bf16_t)(x.b.x*s); lo[5]=(bf16_t)(x.b.y*s); lo[6]=(bf16_t)(x.b.z*s); lo[7]=(bf16_t)(x.b.w*s);
    hi[0]=(bf16_t)(x.c.x*s); hi[1]=(bf16_t)(x.c.y*s); hi[2]=(bf16_t)(x.c.z*s); hi[3]=(bf16_t)(x.c.w*s);
    hi[4]=(bf16_t)(x.d.x*s); hi[5]=(bf16_t)(x.d.y*s); hi[6]=(bf16_t)(x.d.z*s); hi[7]=(bf16_t)(x.d.w*s);
    *(v8bf*)(dst)     = lo;
    *(v8bf*)(dst + 8) = hi;
}

__device__ inline v16bf frag_cat(const bf16_t* p0, const bf16_t* p1) {
    v8bf lo = *(const v8bf*)p0;
    v8bf hi = *(const v8bf*)p1;
    v16bf r;
#pragma unroll
    for (int t = 0; t < 8; ++t) { r[t] = lo[t]; r[t + 8] = hi[t]; }
    return r;
}

__global__ __launch_bounds__(256)
void k_gemm(const float* __restrict__ F, const float* __restrict__ Bd,
            const float* __restrict__ inv_f, const float* __restrict__ inv_b,
            float* __restrict__ rowsum, float* __restrict__ colsum) {
    // double-buffered: 80B row stride = 16B aligned, 16 rows conflict-free
    __shared__ bf16_t As[2][128][40];
    __shared__ bf16_t Bs[2][128][40];

    const int t    = threadIdx.x;
    const int lane = t & 31;
    const int w    = t >> 5;
    const int wr   = w >> 2;            // 0..1  -> 64-row strip
    const int wc   = w & 3;             // 0..3  -> 32-col strip
    const int i0   = blockIdx.y * 128;
    const int j0   = blockIdx.x * 128;

    // staging assignment: thread -> (row, 16-wide k chunk)
    const int srow = t >> 1;
    const int skk  = (t & 1) * 16;
    const float sA = inv_f[i0 + srow];
    const float sB = inv_b[j0 + srow];
    const float* gA = F  + (size_t)(i0 + srow) * D_K + skk;
    const float* gB = Bd + (size_t)(j0 + srow) * D_K + skk;

    v8f acc[4][2];
#pragma unroll
    for (int m = 0; m < 4; ++m)
#pragma unroll
        for (int n = 0; n < 2; ++n)
#pragma unroll
            for (int v = 0; v < 8; ++v) acc[m][n][v] = 0.f;

    const int lr = lane & 15;
    const int hm = lane >> 4;

    // fragment base pointers (per-buffer offset added in loop)
    const bf16_t* apB0 = &As[0][wr * 64 + lr][hm * 8];
    const bf16_t* bpB0 = &Bs[0][wc * 32 + lr][hm * 8];
    const int bufStrideA = (int)(&As[1][0][0] - &As[0][0][0]);   // elements
    const int bufStrideB = (int)(&Bs[1][0][0] - &Bs[0][0][0]);

    // preload K-step 0 into buffer 0
    {
        F16x xa = ld16(gA);
        F16x xb = ld16(gB);
        st16(&As[0][srow][skk], xa, sA);
        st16(&Bs[0][srow][skk], xb, sB);
    }
    __syncthreads();

    const int NT = D_K / 32;
    for (int kt = 0; kt < NT; ++kt) {
        const int cur = kt & 1;
        const int nxt = cur ^ 1;

        // ---- stage tile kt+1 into buffer nxt FIRST (safe: nxt's last readers
        //      were in iteration kt-1, fenced by that iteration's barrier).
        //      No registers held across the WMMA section.
        if (kt + 1 < NT) {
            F16x xa = ld16(gA + (kt + 1) * 32);
            F16x xb = ld16(gB + (kt + 1) * 32);
            st16(&As[nxt][srow][skk], xa, sA);   // waits A loads; B still in flight
            st16(&Bs[nxt][srow][skk], xb, sB);
        }

        // ---- fragments of tile kt from buffer cur: issue ALL ds_loads, then WMMAs
        const bf16_t* ab = apB0 + cur * bufStrideA;
        const bf16_t* bb = bpB0 + cur * bufStrideB;
        // B: col N = lane&15; lanes 0-15: K {0..7,16..23}; lanes 16-31: K {8..15,24..31}
        v16bf b0 = frag_cat(bb,            bb + 16);
        v16bf b1 = frag_cat(bb + 16 * 40,  bb + 16 * 40 + 16);
        // A: row M = lane&15; lanes 0-15: K {0..7,16..23}; lanes 16-31: K {8..15,24..31}
        v16bf a0 = frag_cat(ab,            ab + 16);
        v16bf a1 = frag_cat(ab + 16 * 40,  ab + 16 * 40 + 16);
        v16bf a2 = frag_cat(ab + 32 * 40,  ab + 32 * 40 + 16);
        v16bf a3 = frag_cat(ab + 48 * 40,  ab + 48 * 40 + 16);

        acc[0][0] = __builtin_amdgcn_wmma_f32_16x16x32_bf16(false, a0, false, b0, (short)0, acc[0][0], false, false);
        acc[0][1] = __builtin_amdgcn_wmma_f32_16x16x32_bf16(false, a0, false, b1, (short)0, acc[0][1], false, false);
        acc[1][0] = __builtin_amdgcn_wmma_f32_16x16x32_bf16(false, a1, false, b0, (short)0, acc[1][0], false, false);
        acc[1][1] = __builtin_amdgcn_wmma_f32_16x16x32_bf16(false, a1, false, b1, (short)0, acc[1][1], false, false);
        acc[2][0] = __builtin_amdgcn_wmma_f32_16x16x32_bf16(false, a2, false, b0, (short)0, acc[2][0], false, false);
        acc[2][1] = __builtin_amdgcn_wmma_f32_16x16x32_bf16(false, a2, false, b1, (short)0, acc[2][1], false, false);
        acc[3][0] = __builtin_amdgcn_wmma_f32_16x16x32_bf16(false, a3, false, b0, (short)0, acc[3][0], false, false);
        acc[3][1] = __builtin_amdgcn_wmma_f32_16x16x32_bf16(false, a3, false, b1, (short)0, acc[3][1], false, false);

        __syncthreads();
    }

    // epilogue: e = exp(sim - 1/TEMP) = exp((dot-1)/TEMP); accumulate row/col sums
#pragma unroll
    for (int m = 0; m < 4; ++m)
#pragma unroll
        for (int n = 0; n < 2; ++n)
#pragma unroll
            for (int v = 0; v < 8; ++v)
                acc[m][n][v] = __expf((acc[m][n][v] - 1.0f) * INV_T);

#pragma unroll
    for (int n = 0; n < 2; ++n) {
        float cs = 0.f;
#pragma unroll
        for (int m = 0; m < 4; ++m)
#pragma unroll
            for (int v = 0; v < 8; ++v) cs += acc[m][n][v];
        atomicAdd(&colsum[j0 + wc * 32 + n * 16 + lr], cs);
    }
#pragma unroll
    for (int m = 0; m < 4; ++m) {
#pragma unroll
        for (int v = 0; v < 8; ++v) {
            float rs = acc[m][0][v] + acc[m][1][v];
            rs += __shfl_xor(rs, 1);
            rs += __shfl_xor(rs, 2);
            rs += __shfl_xor(rs, 4);
            rs += __shfl_xor(rs, 8);   // sum over 16 lanes of each half
            if (lr == v)
                atomicAdd(&rowsum[i0 + wr * 64 + m * 16 + hm * 8 + v], rs);
        }
    }
}

// ---------------------------------------------------------------------------
// K3a: triplet mining (first positive, k-th negative, validity)
// ---------------------------------------------------------------------------
__global__ __launch_bounds__(256)
void k_mine(const int* __restrict__ labels, const int* __restrict__ neg_rand,
            int* __restrict__ pos_idx, int* __restrict__ neg_idx,
            float* __restrict__ validf) {
    __shared__ int lab[B_N];
    for (int j = threadIdx.x; j < B_N; j += 256) lab[j] = labels[j];
    __syncthreads();
    const int i = blockIdx.x * 256 + threadIdx.x;
    const int li = lab[i];
    int npos = 0, nneg = 0, fpos = -1;
    for (int j = 0; j < B_N; ++j) {
        if (lab[j] == li) {
            if (j != i) { if (fpos < 0) fpos = j; ++npos; }
        } else {
            ++nneg;
        }
    }
    const int nn = (nneg > 0) ? nneg : 1;
    const int k  = neg_rand[i] % nn;
    int cnt = 0, nidx = 0;
    for (int j = 0; j < B_N; ++j) {
        if (lab[j] != li) {
            if (cnt == k) { nidx = j; break; }
            ++cnt;
        }
    }
    pos_idx[i] = (fpos < 0) ? 0 : fpos;   // argmax of all-false mask == 0
    neg_idx[i] = nidx;
    validf[i]  = (npos >= 1 && nneg >= 1) ? 1.0f : 0.0f;
}

// ---------------------------------------------------------------------------
// K3b: triplet distances + hinge, accumulate sum and count
// ---------------------------------------------------------------------------
__global__ __launch_bounds__(256)
void k_trip(const float* __restrict__ F, const float* __restrict__ Bd,
            const int* __restrict__ pos_idx, const int* __restrict__ neg_idx,
            const float* __restrict__ validf, float* __restrict__ scal) {
    const int i = blockIdx.x;
    const int t = threadIdx.x;
    const float* a = F  + (size_t)i * D_K;
    const float* p = Bd + (size_t)pos_idx[i] * D_K;
    const float* n = Bd + (size_t)neg_idx[i] * D_K;
    float sp = 0.f, sn = 0.f;
    for (int k = t * 4; k < D_K; k += 1024) {
        float4 av = *(const float4*)(a + k);
        float4 pv = *(const float4*)(p + k);
        float4 nv = *(const float4*)(n + k);
        float d;
        d = av.x - pv.x + EPSV; sp += d * d;
        d = av.y - pv.y + EPSV; sp += d * d;
        d = av.z - pv.z + EPSV; sp += d * d;
        d = av.w - pv.w + EPSV; sp += d * d;
        d = av.x - nv.x + EPSV; sn += d * d;
        d = av.y - nv.y + EPSV; sn += d * d;
        d = av.z - nv.z + EPSV; sn += d * d;
        d = av.w - nv.w + EPSV; sn += d * d;
    }
    __shared__ float r0[256], r1[256];
    r0[t] = sp; r1[t] = sn;
    __syncthreads();
    for (int s = 128; s > 0; s >>= 1) {
        if (t < s) { r0[t] += r0[t+s]; r1[t] += r1[t+s]; }
        __syncthreads();
    }
    if (t == 0) {
        float tl = fmaxf(sqrtf(r0[0]) - sqrtf(r1[0]) + MARGIN, 0.0f);
        if (validf[i] > 0.5f) {
            atomicAdd(&scal[0], tl);
            atomicAdd(&scal[1], 1.0f);
        }
    }
}

// ---------------------------------------------------------------------------
// K4: finalize losses
// ---------------------------------------------------------------------------
__global__ __launch_bounds__(256)
void k_final(const float* __restrict__ diag, const float* __restrict__ rowsum,
             const float* __restrict__ colsum, const float* __restrict__ scal,
             float* __restrict__ out) {
    const int t = threadIdx.x;
    float s1 = 0.f, s2 = 0.f;
    for (int i = t; i < B_N; i += 256) {
        float d = (diag[i] - 1.0f) * INV_T;   // sim_ii - M
        s1 += d - logf(rowsum[i]);
        s2 += d - logf(colsum[i]);
    }
    __shared__ float r0[256], r1[256];
    r0[t] = s1; r1[t] = s2;
    __syncthreads();
    for (int s = 128; s > 0; s >>= 1) {
        if (t < s) { r0[t] += r0[t+s]; r1[t] += r1[t+s]; }
        __syncthreads();
    }
    if (t == 0) {
        float infonce  = -(r0[0] + r1[0]) / (2.0f * (float)B_N);
        float cnt      = scal[1];
        float identity = (cnt > 0.f) ? (scal[0] / cnt) : 0.f;
        out[0] = infonce + identity;
        out[1] = infonce;
        out[2] = identity;
    }
}

// ---------------------------------------------------------------------------
extern "C" void kernel_launch(void* const* d_in, const int* in_sizes, int n_in,
                              void* d_out, int out_size, void* d_ws, size_t ws_size,
                              hipStream_t stream) {
    const float* F      = (const float*)d_in[0];
    const float* Bd     = (const float*)d_in[1];
    const int*   labels = (const int*)d_in[2];
    const int*   negr   = (const int*)d_in[3];
    float* out = (float*)d_out;

    float* ws     = (float*)d_ws;
    float* inv_f  = ws;                 // 4096
    float* inv_b  = ws + 4096;          // 4096
    float* diag   = ws + 8192;          // 4096
    float* rowsum = ws + 12288;         // 4096
    float* colsum = ws + 16384;         // 4096
    float* scal   = ws + 20480;         // [0]=triplet sum, [1]=valid count
    int*   pos_i  = (int*)(ws + 20544); // 4096
    int*   neg_i  = pos_i + 4096;       // 4096
    float* validf = (float*)(neg_i + 4096); // 4096

    k_prep<<<B_N, 256, 0, stream>>>(F, Bd, inv_f, inv_b, diag, rowsum, colsum, scal);

    dim3 g(B_N / 128, B_N / 128);
    k_gemm<<<g, 256, 0, stream>>>(F, Bd, inv_f, inv_b, rowsum, colsum);

    k_mine<<<B_N / 256, 256, 0, stream>>>(labels, negr, pos_i, neg_i, validf);
    k_trip<<<B_N, 256, 0, stream>>>(F, Bd, pos_i, neg_i, validf, scal);
    k_final<<<1, 256, 0, stream>>>(diag, rowsum, colsum, scal, out);
}
